// MultiSimilarityLoss_29222957482318
// MI455X (gfx1250) — matrix operand order repaired and checked by reading, not compiled, not run on hardware
//
#include <hip/hip_runtime.h>
#include <hip/hip_bf16.h>

typedef __attribute__((ext_vector_type(16))) _Float16 v16h;
typedef __attribute__((ext_vector_type(8)))  _Float16 v8h;
typedef __attribute__((ext_vector_type(8)))  float    v8f;

#define MS_THRESH    0.5f
#define MS_MARGIN    0.1f
#define MS_SCALE_POS 2.0f
#define MS_SCALE_NEG 40.0f
#define MS_EPS       1e-5f

#define NWAVES   32              // 1024 threads/workgroup (wave32)
#define NTHREADS (NWAVES * 32)
#define TPW      8               // 16-wide column tiles per wave -> NWAVES*TPW*16 = 4096 cols
#define TM       16              // rows per workgroup

// ---------------- kernel 0: split f32 feats into f16 hi + f16 lo ----------------
__global__ void ms_split(const float* __restrict__ x, _Float16* __restrict__ hi,
                         _Float16* __restrict__ lo, int n) {
    int i = blockIdx.x * blockDim.x + threadIdx.x;
    if (i < n) {
        float v = x[i];
        _Float16 h = (_Float16)v;
        hi[i] = h;
        lo[i] = (_Float16)(v - (float)h);   // exact residual, rounded to f16
    }
}

// ---------------- kernel 1: fused sim strip (WMMA) + mining + row reductions ----
// One workgroup owns 16 rows of sim. 32 waves each hold 8 register-resident
// 16x16 f32 accumulator tiles covering 128 columns. Requires B == 4096, D % 32 == 0.
__global__ __launch_bounds__(NTHREADS)
void ms_main(const _Float16* __restrict__ hi, const _Float16* __restrict__ lo,
             const int* __restrict__ labels, const int* __restrict__ plabel_num,
             float* __restrict__ partials, int B, int D) {
    const int tid   = threadIdx.x;
    const int wave  = tid >> 5;
    const int lane  = tid & 31;
    const int laneM = lane & 15;   // row (A) / column (B) index within tile
    const int laneS = lane >> 4;   // half-wave selector
    const int r0    = blockIdx.x * TM;

    __shared__ float red0[NWAVES][16];
    __shared__ float red1[NWAVES][16];
    __shared__ float red2[NWAVES][16];
    __shared__ float red3[NWAVES][16];
    __shared__ float thrP[16], thrN[16];
    __shared__ float rowres[16];

    v8f acc[TPW];
#pragma unroll
    for (int t = 0; t < TPW; ++t) { v8f z = {}; acc[t] = z; }

    const int aBase = (r0 + laneM) * D;

    // ---- GEMM: sim[r0:r0+16, wave's 128 cols], hi/lo split => ~f32 accuracy ----
    for (int k0 = 0; k0 < D; k0 += 32) {
        // A fragment (16-bit A 16x32, contiguous k order): lane m+16s holds row m,
        // K = k0+16s+0..15 (matches B) -> two b128 loads
        v8h ah0 = *(const v8h*)(hi + aBase + k0 + 16 * laneS);
        v8h ah1 = *(const v8h*)(hi + aBase + k0 + 16 * laneS + 8);
        v8h al0 = *(const v8h*)(lo + aBase + k0 + 16 * laneS);
        v8h al1 = *(const v8h*)(lo + aBase + k0 + 16 * laneS + 8);
        v16h aHi = __builtin_shufflevector(ah0, ah1, 0,1,2,3,4,5,6,7,8,9,10,11,12,13,14,15);
        v16h aLo = __builtin_shufflevector(al0, al1, 0,1,2,3,4,5,6,7,8,9,10,11,12,13,14,15);
#pragma unroll
        for (int t = 0; t < TPW; ++t) {
            // B fragment (16-bit B 32x16): lane n+16s holds col n, K = k0+16s+0..15
            // B[k][n] = feats[col n][k]  (sim is feats @ feats^T)
            const int bOff = ((wave * TPW + t) * 16 + laneM) * D + k0 + 16 * laneS;
            v16h bHi = *(const v16h*)(hi + bOff);
            v16h bLo = *(const v16h*)(lo + bOff);
            acc[t] = __builtin_amdgcn_wmma_f32_16x16x32_f16(false, aHi, false, bHi,
                                                            (short)0, acc[t], false, false);
            acc[t] = __builtin_amdgcn_wmma_f32_16x16x32_f16(false, aHi, false, bLo,
                                                            (short)0, acc[t], false, false);
            acc[t] = __builtin_amdgcn_wmma_f32_16x16x32_f16(false, aLo, false, bHi,
                                                            (short)0, acc[t], false, false);
        }
    }

    // C/D layout: acc[t][j] = sim[r0 + j + 8*laneS, (wave*TPW+t)*16 + laneM]
    int rlab[8];
#pragma unroll
    for (int j = 0; j < 8; ++j) rlab[j] = labels[r0 + j + 8 * laneS];
    int clab[TPW];
#pragma unroll
    for (int t = 0; t < TPW; ++t) clab[t] = labels[(wave * TPW + t) * 16 + laneM];

    // ---- pass 1: per-row min(pos sim) / max(neg sim) ----
    float mnp[8], mxn[8];
#pragma unroll
    for (int j = 0; j < 8; ++j) { mnp[j] = 3.0e38f; mxn[j] = -3.0e38f; }
#pragma unroll
    for (int t = 0; t < TPW; ++t) {
#pragma unroll
        for (int j = 0; j < 8; ++j) {
            float s = acc[t][j];
            if (rlab[j] == clab[t]) {
                if (s < 1.0f - MS_EPS) mnp[j] = fminf(mnp[j], s);
            } else {
                mxn[j] = fmaxf(mxn[j], s);
            }
        }
    }
#pragma unroll
    for (int j = 0; j < 8; ++j) {
#pragma unroll
        for (int m = 1; m < 16; m <<= 1) {   // butterfly within each 16-lane group
            mnp[j] = fminf(mnp[j], __shfl_xor(mnp[j], m));
            mxn[j] = fmaxf(mxn[j], __shfl_xor(mxn[j], m));
        }
    }
    if (laneM == 0) {
#pragma unroll
        for (int j = 0; j < 8; ++j) {
            red0[wave][j + 8 * laneS] = mnp[j];
            red1[wave][j + 8 * laneS] = mxn[j];
        }
    }
    __syncthreads();
    if (tid < 16) {
        float mn = 3.0e38f, mx = -3.0e38f;
        for (int w = 0; w < NWAVES; ++w) {
            mn = fminf(mn, red0[w][tid]);
            mx = fmaxf(mx, red1[w][tid]);
        }
        thrP[tid] = (mn <  1.0e38f) ? mn : 0.2f;   // min_pos or no-pos fallback
        thrN[tid] = (mx > -1.0e38f) ? mx : 0.8f;   // max_neg or no-neg fallback
    }
    __syncthreads();

    // ---- pass 2: mined pair exp-sums over the same register tiles ----
    float ps[8], ns[8], pcnt[8], ncnt[8];
#pragma unroll
    for (int j = 0; j < 8; ++j) { ps[j] = 0.f; ns[j] = 0.f; pcnt[j] = 0.f; ncnt[j] = 0.f; }
#pragma unroll
    for (int t = 0; t < TPW; ++t) {
#pragma unroll
        for (int j = 0; j < 8; ++j) {
            float s = acc[t][j];
            const int row = j + 8 * laneS;
            if (rlab[j] == clab[t]) {
                if ((s < 1.0f - MS_EPS) && (s - MS_MARGIN < thrN[row])) {
                    ps[j] += __expf(-MS_SCALE_POS * (s - MS_THRESH));
                    pcnt[j] += 1.0f;
                }
            } else {
                if (s + MS_MARGIN > thrP[row]) {
                    ns[j] += __expf(MS_SCALE_NEG * (s - MS_THRESH));
                    ncnt[j] += 1.0f;
                }
            }
        }
    }
#pragma unroll
    for (int j = 0; j < 8; ++j) {
#pragma unroll
        for (int m = 1; m < 16; m <<= 1) {
            ps[j]   += __shfl_xor(ps[j],   m);
            ns[j]   += __shfl_xor(ns[j],   m);
            pcnt[j] += __shfl_xor(pcnt[j], m);
            ncnt[j] += __shfl_xor(ncnt[j], m);
        }
    }
    if (laneM == 0) {
#pragma unroll
        for (int j = 0; j < 8; ++j) {
            const int row = j + 8 * laneS;
            red0[wave][row] = ps[j];
            red1[wave][row] = ns[j];
            red2[wave][row] = pcnt[j];
            red3[wave][row] = ncnt[j];
        }
    }
    __syncthreads();
    if (tid < 16) {
        float sps = 0.f, sns = 0.f, spc = 0.f, snc = 0.f;
        for (int w = 0; w < NWAVES; ++w) {
            sps += red0[w][tid]; sns += red1[w][tid];
            spc += red2[w][tid]; snc += red3[w][tid];
        }
        const int label_num = *plabel_num;
        const int grow = r0 + tid;
        const bool valid = (spc > 0.f) && (snc > 0.f) && (grow < B - label_num);
        rowres[tid] = valid ? ((1.0f / MS_SCALE_POS) * log1pf(sps) +
                               (1.0f / MS_SCALE_NEG) * log1pf(sns))
                            : 0.0f;
    }
    __syncthreads();
    if (tid == 0) {
        float s = 0.f;
        for (int i = 0; i < 16; ++i) s += rowres[i];
        partials[blockIdx.x] = s;           // plain store -> deterministic
    }
}

// ---------------- kernel 2: deterministic final reduction ----------------
__global__ void ms_finalize(const float* __restrict__ partials, int nb,
                            float* __restrict__ out, int B) {
    __shared__ float s[256];
    const int t = threadIdx.x;
    float v = 0.f;
    for (int i = t; i < nb; i += 256) v += partials[i];
    s[t] = v;
    __syncthreads();
    for (int off = 128; off > 0; off >>= 1) {
        if (t < off) s[t] += s[t + off];
        __syncthreads();
    }
    if (t == 0) out[0] = s[0] / (float)B;
}

extern "C" void kernel_launch(void* const* d_in, const int* in_sizes, int n_in,
                              void* d_out, int out_size, void* d_ws, size_t ws_size,
                              hipStream_t stream) {
    (void)n_in; (void)out_size; (void)ws_size;
    const float* feats      = (const float*)d_in[0];
    const int*   labels     = (const int*)d_in[1];   // JAX x64 off -> int32
    const int*   plabel_num = (const int*)d_in[2];

    const int B = in_sizes[1];           // 4096
    const int D = in_sizes[0] / B;       // 512
    const int n = B * D;

    // workspace: [ hi f16 (n) | lo f16 (n) | partials f32 (B/16) ]  ~= 8 MB + 1 KB
    _Float16* hi = (_Float16*)d_ws;
    _Float16* lo = hi + n;
    float* partials = (float*)((char*)d_ws + 2 * (size_t)n * sizeof(_Float16));

    ms_split<<<(n + 255) / 256, 256, 0, stream>>>(feats, hi, lo, n);

    const int nblocks = B / TM;          // 256 row strips
    ms_main<<<nblocks, NTHREADS, 0, stream>>>(hi, lo, labels, plabel_num, partials, B, D);

    ms_finalize<<<1, 256, 0, stream>>>(partials, nblocks, (float*)d_out, B);
}
